// Spatial_Feature_Layer_4105988735551
// MI455X (gfx1250) — matrix operand order repaired and checked by reading, not compiled, not run on hardware
//
#include <hip/hip_runtime.h>
#include <stdint.h>

// Problem constants
#define NNODE 512
#define DIM   128
#define DFF   2048
#define LN_EPS 1e-5f

typedef __attribute__((ext_vector_type(16))) __bf16 v16bf;
typedef __attribute__((ext_vector_type(8)))  float  v8f;

// ---------- bf16 helpers ----------
__device__ __forceinline__ unsigned short f2bf(float x) {
  unsigned int u = __float_as_uint(x);
  u = u + 0x7FFFu + ((u >> 16) & 1u);      // round-to-nearest-even
  return (unsigned short)(u >> 16);
}
__device__ __forceinline__ float bf2f(unsigned short s) {
  return __uint_as_float(((unsigned int)s) << 16);
}
__device__ __forceinline__ unsigned int pack2(float a, float b) {
  return (unsigned int)f2bf(a) | ((unsigned int)f2bf(b) << 16);
}

// ---------- WMMA ----------
__device__ __forceinline__ v8f wmma_bf16(v16bf a, v16bf b, v8f c) {
  // D = A(16x32 bf16) * B(32x16 bf16) + C(16x16 f32)
  return __builtin_amdgcn_wmma_f32_16x16x32_bf16(false, a, false, b, (short)0, c,
                                                 false, false);
}

// A fragment from LDS tile stored row-major [rows][128] bf16.
// Wave32 layout: lane L holds row rowb+(L&15); elems 0..7 -> K=kb+8*half+0..7,
// elems 8..15 -> K=kb+16+8*half+0..7  (two ds_read_b128).
__device__ __forceinline__ v16bf load_frag_A(const unsigned short* tile, int rowb,
                                             int kb, int lane) {
  int half = lane >> 4;
  const unsigned short* p = tile + (rowb + (lane & 15)) * 128 + kb + 8 * half;
  union { v16bf v; uint4 q[2]; } u;
  u.q[0] = *(const uint4*)(p);
  u.q[1] = *(const uint4*)(p + 16);
  return u.v;
}
// B fragment from LDS weight stored TRANSPOSED [n(0..127)][k(0..127)] bf16.
// Lane L: col n0+(L&15); elems 0..7 -> K = kb + 8*half + 0..7, elems 8..15 -> K = kb + 16 + 8*half + 0..7.
__device__ __forceinline__ v16bf load_frag_B(const unsigned short* wt, int n0,
                                             int kb, int lane) {
  const unsigned short* p = wt + (n0 + (lane & 15)) * 128 + kb + 8 * (lane >> 4);
  union { v16bf v; uint4 q[2]; } u;
  u.q[0] = *(const uint4*)(p);
  u.q[1] = *(const uint4*)(p + 16);
  return u.v;
}

// ---------- reductions ----------
// Sum across each 16-lane half of the wave (C-layout: one full output row lives
// in one 16-lane group x 8 col tiles), so LayerNorm needs no LDS round-trip.
__device__ __forceinline__ float hsum16(float v) {
  v += __shfl_xor(v, 1, 32);
  v += __shfl_xor(v, 2, 32);
  v += __shfl_xor(v, 4, 32);
  v += __shfl_xor(v, 8, 32);
  return v;
}
__device__ __forceinline__ float wsum32(float v) {
  v += __shfl_xor(v, 1, 32);
  v += __shfl_xor(v, 2, 32);
  v += __shfl_xor(v, 4, 32);
  v += __shfl_xor(v, 8, 32);
  v += __shfl_xor(v, 16, 32);
  return v;
}
__device__ __forceinline__ float wmax32(float v) {
  v = fmaxf(v, __shfl_xor(v, 1, 32));
  v = fmaxf(v, __shfl_xor(v, 2, 32));
  v = fmaxf(v, __shfl_xor(v, 4, 32));
  v = fmaxf(v, __shfl_xor(v, 8, 32));
  v = fmaxf(v, __shfl_xor(v, 16, 32));
  return v;
}

// =====================================================================
// K0: s = node@Ws, t = node@Wt, q = (node@Wq + bq)/sqrt(HD)   (512x128 each)
// =====================================================================
__global__ void k0_precompute(const float* __restrict__ node,
                              const float* __restrict__ W_mem,
                              const float* __restrict__ Wq,
                              const float* __restrict__ bq,
                              float* __restrict__ s_buf,
                              float* __restrict__ t_buf,
                              float* __restrict__ q_buf) {
  __shared__ float nrow[128];
  int r = blockIdx.x, c = threadIdx.x;
  nrow[c] = node[r * 128 + c];
  __syncthreads();
  float ss = 0.f, tt = 0.f, qq = bq[c];
  for (int k = 0; k < 128; ++k) {
    float nv = nrow[k];
    ss += nv * W_mem[(128 + k) * 128 + c];   // Ws rows of W_mem
    tt += nv * W_mem[(256 + k) * 128 + c];   // Wt rows of W_mem
    qq += nv * Wq[k * 128 + c];
  }
  s_buf[r * 128 + c] = ss;
  t_buf[r * 128 + c] = tt;
  q_buf[r * 128 + c] = qq * 0.25f;           // 1/sqrt(16)
}

// =====================================================================
// KA: one pass over edge. Per 128-row tile:
//   memory = relu(ln(edge@We + s[j] + t[i] + b_mem))       (LDS only)
//   edge_out = ln(edge + relu(ln(memory@Wpe + b_pe)))      -> HBM
//   kT[b][s] = bf16(memory@Wk + bk), vT likewise           -> HBM (bf16)
// All four GEMMs via v_wmma_f32_16x16x32_bf16; LN via lane shuffles.
// =====================================================================
__global__ void __launch_bounds__(256)
ka_edge_fused(const float* __restrict__ edge,
              const float* __restrict__ We,  const float* __restrict__ Wpe,
              const float* __restrict__ Wk,  const float* __restrict__ Wv,
              const float* __restrict__ b_mem, const float* __restrict__ g_mem,
              const float* __restrict__ be_mem,
              const float* __restrict__ b_pe,  const float* __restrict__ g_pe,
              const float* __restrict__ be_pe,
              const float* __restrict__ g_ne,  const float* __restrict__ be_ne,
              const float* __restrict__ bk,    const float* __restrict__ bv,
              const float* __restrict__ s_buf, const float* __restrict__ t_buf,
              unsigned short* __restrict__ kT, unsigned short* __restrict__ vT,
              float* __restrict__ edge_out) {
  extern __shared__ char smem[];
  unsigned short* Atile = (unsigned short*)smem;            // edge tile bf16 128x128
  unsigned short* Mtile = (unsigned short*)(smem + 32768);  // memory tile bf16
  unsigned short* Wt    = (unsigned short*)(smem + 65536);  // transposed weight bf16
  float* cv             = (float*)(smem + 98304);           // 11*128 const vectors

  const int tid  = threadIdx.x;
  const int lane = tid & 31;
  const int wave = tid >> 5;
  const int r0   = blockIdx.x * 128;     // flat row base (i*512 + j)
  const int ibase = r0 >> 9;             // constant i for this WG
  const int jbase = r0 & (NNODE - 1);    // j base

  if (tid < 128) {
    cv[tid]          = b_mem[tid]; cv[128 + tid] = g_mem[tid]; cv[256 + tid] = be_mem[tid];
    cv[384 + tid]    = b_pe[tid];  cv[512 + tid] = g_pe[tid];  cv[640 + tid] = be_pe[tid];
    cv[768 + tid]    = g_ne[tid];  cv[896 + tid] = be_ne[tid];
    cv[1024 + tid]   = bk[tid];    cv[1152 + tid] = bv[tid];
    cv[1280 + tid]   = t_buf[(size_t)ibase * 128 + tid];
  }
  // stage edge tile -> bf16
  const float4* esrc = (const float4*)edge + (size_t)r0 * 32;
  uint2* adst = (uint2*)Atile;
  for (int f = tid; f < 4096; f += 256) {
    float4 e = esrc[f];
    adst[f] = make_uint2(pack2(e.x, e.y), pack2(e.z, e.w));
  }
  __syncthreads();

  const int rowb = wave * 16;
  const int col0 = lane & 15, half = lane >> 4;
  v8f accs[8];

  // ---------------- GEMM1: edge @ We ----------------
  for (int f = tid; f < 16384; f += 256) {
    int k = f >> 7, n = f & 127;
    Wt[n * 128 + k] = f2bf(We[f]);
  }
  __syncthreads();
  {
    v16bf afr[4];
#pragma unroll
    for (int ks = 0; ks < 4; ++ks) afr[ks] = load_frag_A(Atile, rowb, ks * 32, lane);
#pragma unroll
    for (int ct = 0; ct < 8; ++ct) {
      v8f acc = {0, 0, 0, 0, 0, 0, 0, 0};
#pragma unroll
      for (int ks = 0; ks < 4; ++ks)
        acc = wmma_bf16(afr[ks], load_frag_B(Wt, ct * 16, ks * 32, lane), acc);
      accs[ct] = acc;
    }
  }
  // epilogue 1: + b_mem + s[j] + t[i], LN, relu -> Mtile (bf16)
#pragma unroll
  for (int v = 0; v < 8; ++v) {
    int m = rowb + v + 8 * half;
    const float* srow = s_buf + (size_t)(jbase + m) * 128;
    float xs[8], sum = 0.f;
#pragma unroll
    for (int ct = 0; ct < 8; ++ct) {
      int col = ct * 16 + col0;
      float x = accs[ct][v] + cv[col] + cv[1280 + col] + srow[col];
      xs[ct] = x; sum += x;
    }
    float mean = hsum16(sum) * (1.f / 128.f);
    float vs = 0.f;
#pragma unroll
    for (int ct = 0; ct < 8; ++ct) { float d = xs[ct] - mean; vs += d * d; }
    float rstd = rsqrtf(hsum16(vs) * (1.f / 128.f) + LN_EPS);
#pragma unroll
    for (int ct = 0; ct < 8; ++ct) {
      int col = ct * 16 + col0;
      float y = (xs[ct] - mean) * rstd * cv[128 + col] + cv[256 + col];
      Mtile[m * 128 + col] = f2bf(fmaxf(y, 0.f));
    }
  }
  __syncthreads();

  // ---------------- GEMM2: memory @ Wpe -> edge_out ----------------
  for (int f = tid; f < 16384; f += 256) {
    int k = f >> 7, n = f & 127;
    Wt[n * 128 + k] = f2bf(Wpe[f]);
  }
  __syncthreads();
  v16bf mfr[4];
#pragma unroll
  for (int ks = 0; ks < 4; ++ks) mfr[ks] = load_frag_A(Mtile, rowb, ks * 32, lane);
#pragma unroll
  for (int ct = 0; ct < 8; ++ct) {
    v8f acc = {0, 0, 0, 0, 0, 0, 0, 0};
#pragma unroll
    for (int ks = 0; ks < 4; ++ks)
      acc = wmma_bf16(mfr[ks], load_frag_B(Wt, ct * 16, ks * 32, lane), acc);
    accs[ct] = acc;
  }
#pragma unroll
  for (int v = 0; v < 8; ++v) {
    int m = rowb + v + 8 * half;
    float xs[8], sum = 0.f;
#pragma unroll
    for (int ct = 0; ct < 8; ++ct) {
      int col = ct * 16 + col0;
      float x = accs[ct][v] + cv[384 + col];
      xs[ct] = x; sum += x;
    }
    float mean = hsum16(sum) * (1.f / 128.f);
    float vs = 0.f;
#pragma unroll
    for (int ct = 0; ct < 8; ++ct) { float d = xs[ct] - mean; vs += d * d; }
    float rstd = rsqrtf(hsum16(vs) * (1.f / 128.f) + LN_EPS);
    float es[8], sum2 = 0.f;
#pragma unroll
    for (int ct = 0; ct < 8; ++ct) {
      int col = ct * 16 + col0;
      float pe = fmaxf((xs[ct] - mean) * rstd * cv[512 + col] + cv[640 + col], 0.f);
      float e = bf2f(Atile[m * 128 + col]) + pe;
      es[ct] = e; sum2 += e;
    }
    float mean2 = hsum16(sum2) * (1.f / 128.f);
    float vs2 = 0.f;
#pragma unroll
    for (int ct = 0; ct < 8; ++ct) { float d = es[ct] - mean2; vs2 += d * d; }
    float rstd2 = rsqrtf(hsum16(vs2) * (1.f / 128.f) + LN_EPS);
#pragma unroll
    for (int ct = 0; ct < 8; ++ct) {
      int col = ct * 16 + col0;
      edge_out[(size_t)(r0 + m) * 128 + col] =
          (es[ct] - mean2) * rstd2 * cv[768 + col] + cv[896 + col];
    }
  }
  __syncthreads();

  // ---------------- GEMM3: memory @ Wk -> kT[b][s] bf16 ----------------
  for (int f = tid; f < 16384; f += 256) {
    int k = f >> 7, n = f & 127;
    Wt[n * 128 + k] = f2bf(Wk[f]);
  }
  __syncthreads();
#pragma unroll
  for (int ct = 0; ct < 8; ++ct) {
    v8f acc = {0, 0, 0, 0, 0, 0, 0, 0};
#pragma unroll
    for (int ks = 0; ks < 4; ++ks)
      acc = wmma_bf16(mfr[ks], load_frag_B(Wt, ct * 16, ks * 32, lane), acc);
    accs[ct] = acc;
  }
#pragma unroll
  for (int v = 0; v < 8; ++v) {
    int m = rowb + v + 8 * half;
    size_t base = ((size_t)(jbase + m) * NNODE + ibase) * 128;  // [b=j][s=i]
#pragma unroll
    for (int ct = 0; ct < 8; ++ct) {
      int col = ct * 16 + col0;
      kT[base + col] = f2bf(accs[ct][v] + cv[1024 + col]);
    }
  }
  __syncthreads();

  // ---------------- GEMM4: memory @ Wv -> vT[b][s] bf16 ----------------
  for (int f = tid; f < 16384; f += 256) {
    int k = f >> 7, n = f & 127;
    Wt[n * 128 + k] = f2bf(Wv[f]);
  }
  __syncthreads();
#pragma unroll
  for (int ct = 0; ct < 8; ++ct) {
    v8f acc = {0, 0, 0, 0, 0, 0, 0, 0};
#pragma unroll
    for (int ks = 0; ks < 4; ++ks)
      acc = wmma_bf16(mfr[ks], load_frag_B(Wt, ct * 16, ks * 32, lane), acc);
    accs[ct] = acc;
  }
#pragma unroll
  for (int v = 0; v < 8; ++v) {
    int m = rowb + v + 8 * half;
    size_t base = ((size_t)(jbase + m) * NNODE + ibase) * 128;
#pragma unroll
    for (int ct = 0; ct < 8; ++ct) {
      int col = ct * 16 + col0;
      vT[base + col] = f2bf(accs[ct][v] + cv[1152 + col]);
    }
  }
}

// =====================================================================
// KB1: per-node masked attention + Wo projection + LN2  -> x (f32 + bf16)
// =====================================================================
__global__ void __launch_bounds__(256)
kb1_attention(const float* __restrict__ q_buf,
              const unsigned short* __restrict__ kT,
              const unsigned short* __restrict__ vT,
              const unsigned char* __restrict__ mask,
              const float* __restrict__ Wo, const float* __restrict__ bo,
              const float* __restrict__ node,
              const float* __restrict__ g2, const float* __restrict__ beta2,
              float* __restrict__ x_f32, unsigned short* __restrict__ x_bf16) {
  __shared__ float qs[128];
  __shared__ float sc[8 * 512];
  __shared__ float ctx[128];
  __shared__ float xv[128];
  __shared__ float red[8];
  const int tid = threadIdx.x, lane = tid & 31, wave = tid >> 5;
  const int b = blockIdx.x;

  if (tid < 128) qs[tid] = q_buf[b * 128 + tid];
  __syncthreads();

  for (int s = tid; s < 512; s += 256) {
    const unsigned short* kr = kT + ((size_t)b * 512 + s) * 128;
    float hacc[8] = {0, 0, 0, 0, 0, 0, 0, 0};
    for (int c = 0; c < 128; ++c) hacc[c >> 4] += qs[c] * bf2f(kr[c]);
    bool mk = mask[(size_t)b * 512 + s] != 0;
#pragma unroll
    for (int h = 0; h < 8; ++h) sc[h * 512 + s] = mk ? -3.0e38f : hacc[h];
  }
  __syncthreads();

  { // softmax: wave w owns head w
    int h = wave;
    float mx = -3.4e38f;
    for (int s = lane; s < 512; s += 32) mx = fmaxf(mx, sc[h * 512 + s]);
    mx = wmax32(mx);
    float sm = 0.f;
    for (int s = lane; s < 512; s += 32) {
      float e = __expf(sc[h * 512 + s] - mx);
      sc[h * 512 + s] = e; sm += e;
    }
    sm = wsum32(sm);
    float inv = 1.f / sm;
    for (int s = lane; s < 512; s += 32) sc[h * 512 + s] *= inv;
  }
  __syncthreads();

  if (tid < 128) {  // ctx
    int c = tid, h = c >> 4;
    const unsigned short* vb = vT + (size_t)b * 512 * 128 + c;
    float a = 0.f;
    for (int s = 0; s < 512; ++s) a += sc[h * 512 + s] * bf2f(vb[(size_t)s * 128]);
    ctx[c] = a;
  }
  __syncthreads();
  if (tid < 128) {  // ctx @ Wo + bo + node
    int c = tid;
    float a = bo[c];
    for (int k = 0; k < 128; ++k) a += ctx[k] * Wo[k * 128 + c];
    xv[c] = a + node[b * 128 + c];
  }
  __syncthreads();

  // block LayerNorm over xv[0..127]
  float v0 = (tid < 128) ? xv[tid] : 0.f;
  float w = wsum32(v0);
  if (lane == 0) red[wave] = w;
  __syncthreads();
  float tot = 0.f;
#pragma unroll
  for (int i = 0; i < 8; ++i) tot += red[i];
  float mean = tot * (1.f / 128.f);
  __syncthreads();
  float d = (tid < 128) ? (v0 - mean) : 0.f;
  float w2 = wsum32(d * d);
  if (lane == 0) red[wave] = w2;
  __syncthreads();
  float tot2 = 0.f;
#pragma unroll
  for (int i = 0; i < 8; ++i) tot2 += red[i];
  float rstd = rsqrtf(tot2 * (1.f / 128.f) + LN_EPS);
  if (tid < 128) {
    float y = (v0 - mean) * rstd * g2[tid] + beta2[tid];
    x_f32[b * 128 + tid] = y;
    x_bf16[b * 128 + tid] = f2bf(y);
  }
}

// =====================================================================
// KB2a: h1 = relu(x @ W1 + b1)   M=512 K=128 N=2048, WMMA, bf16 out
// =====================================================================
__global__ void __launch_bounds__(256)
kb2a_ffn1(const unsigned short* __restrict__ x_bf16, const float* __restrict__ W1,
          const float* __restrict__ b1, unsigned short* __restrict__ h1) {
  extern __shared__ char smem[];
  unsigned short* Atile = (unsigned short*)smem;            // 128x128 bf16
  unsigned short* Wt    = (unsigned short*)(smem + 32768);  // 128x128 bf16
  const int tid = threadIdx.x, lane = tid & 31, wave = tid >> 5;
  const int c0 = blockIdx.x * 128, row0 = blockIdx.y * 128;

  for (int f = tid; f < 4096; f += 256)
    ((unsigned int*)Atile)[f] = ((const unsigned int*)x_bf16)[row0 * 64 + f];
  for (int f = tid; f < 16384; f += 256) {
    int k = f >> 7, n = f & 127;
    Wt[n * 128 + k] = f2bf(W1[(size_t)k * 2048 + c0 + n]);
  }
  __syncthreads();

  const int rowb = wave * 16, col0 = lane & 15, half = lane >> 4;
  v16bf afr[4];
#pragma unroll
  for (int ks = 0; ks < 4; ++ks) afr[ks] = load_frag_A(Atile, rowb, ks * 32, lane);
#pragma unroll
  for (int ct = 0; ct < 8; ++ct) {
    v8f acc = {0, 0, 0, 0, 0, 0, 0, 0};
#pragma unroll
    for (int ks = 0; ks < 4; ++ks)
      acc = wmma_bf16(afr[ks], load_frag_B(Wt, ct * 16, ks * 32, lane), acc);
#pragma unroll
    for (int v = 0; v < 8; ++v) {
      int m = rowb + v + 8 * half;
      int col = ct * 16 + col0;
      float y = fmaxf(acc[v] + b1[c0 + col], 0.f);
      h1[(size_t)(row0 + m) * 2048 + c0 + col] = f2bf(y);
    }
  }
}

// =====================================================================
// KB2b: out_x = ln(x + h1 @ W2 + b2)   M=512 K=2048 N=128, WMMA
// =====================================================================
__global__ void __launch_bounds__(256)
kb2b_ffn2(const unsigned short* __restrict__ h1, const float* __restrict__ W2,
          const float* __restrict__ b2, const float* __restrict__ x_f32,
          const float* __restrict__ g3, const float* __restrict__ beta3,
          float* __restrict__ out_x) {
  extern __shared__ char smem[];
  unsigned short* Atile = (unsigned short*)smem;
  unsigned short* Wt    = (unsigned short*)(smem + 32768);
  const int tid = threadIdx.x, lane = tid & 31, wave = tid >> 5;
  const int row0 = blockIdx.x * 128;
  const int rowb = wave * 16, col0 = lane & 15, half = lane >> 4;

  v8f accs[8];
#pragma unroll
  for (int ct = 0; ct < 8; ++ct) accs[ct] = (v8f){0, 0, 0, 0, 0, 0, 0, 0};

  for (int kc = 0; kc < 16; ++kc) {
    __syncthreads();
    for (int f = tid; f < 4096; f += 256) {
      int rl = f >> 6, kk = f & 63;
      ((unsigned int*)Atile)[rl * 64 + kk] =
          ((const unsigned int*)(h1 + (size_t)(row0 + rl) * 2048 + kc * 128))[kk];
    }
    for (int f = tid; f < 16384; f += 256) {
      int k = f >> 7, n = f & 127;
      Wt[n * 128 + k] = f2bf(W2[(size_t)(kc * 128 + k) * 128 + n]);
    }
    __syncthreads();
    v16bf afr[4];
#pragma unroll
    for (int ks = 0; ks < 4; ++ks) afr[ks] = load_frag_A(Atile, rowb, ks * 32, lane);
#pragma unroll
    for (int ct = 0; ct < 8; ++ct) {
#pragma unroll
      for (int ks = 0; ks < 4; ++ks)
        accs[ct] = wmma_bf16(afr[ks], load_frag_B(Wt, ct * 16, ks * 32, lane), accs[ct]);
    }
  }

  // epilogue: +b2 +x residual, LN(g3,beta3)
#pragma unroll
  for (int v = 0; v < 8; ++v) {
    int m = rowb + v + 8 * half;
    float xs[8], sum = 0.f;
#pragma unroll
    for (int ct = 0; ct < 8; ++ct) {
      int col = ct * 16 + col0;
      float x = accs[ct][v] + b2[col] + x_f32[(size_t)(row0 + m) * 128 + col];
      xs[ct] = x; sum += x;
    }
    float mean = hsum16(sum) * (1.f / 128.f);
    float vs = 0.f;
#pragma unroll
    for (int ct = 0; ct < 8; ++ct) { float d = xs[ct] - mean; vs += d * d; }
    float rstd = rsqrtf(hsum16(vs) * (1.f / 128.f) + LN_EPS);
#pragma unroll
    for (int ct = 0; ct < 8; ++ct) {
      int col = ct * 16 + col0;
      out_x[(size_t)(row0 + m) * 128 + col] =
          (xs[ct] - mean) * rstd * g3[col] + beta3[col];
    }
  }
}

// =====================================================================
extern "C" void kernel_launch(void* const* d_in, const int* in_sizes, int n_in,
                              void* d_out, int out_size, void* d_ws, size_t ws_size,
                              hipStream_t stream) {
  const float* node    = (const float*)d_in[0];
  const float* edge    = (const float*)d_in[1];
  const unsigned char* emask = (const unsigned char*)d_in[2];
  const float* W_mem   = (const float*)d_in[3];   // (384,128): We|Ws|Wt
  const float* b_mem   = (const float*)d_in[4];
  const float* g_mem   = (const float*)d_in[5];
  const float* be_mem  = (const float*)d_in[6];
  const float* W_pe    = (const float*)d_in[7];
  const float* b_pe    = (const float*)d_in[8];
  const float* g_pe    = (const float*)d_in[9];
  const float* be_pe   = (const float*)d_in[10];
  const float* g_ne    = (const float*)d_in[11];
  const float* be_ne   = (const float*)d_in[12];
  const float* Wq      = (const float*)d_in[13];
  const float* bq      = (const float*)d_in[14];
  const float* Wk      = (const float*)d_in[15];
  const float* bk      = (const float*)d_in[16];
  const float* Wv      = (const float*)d_in[17];
  const float* bv      = (const float*)d_in[18];
  const float* Wo      = (const float*)d_in[19];
  const float* bo      = (const float*)d_in[20];
  const float* W1      = (const float*)d_in[21];
  const float* b1      = (const float*)d_in[22];
  const float* W2      = (const float*)d_in[23];
  const float* b2      = (const float*)d_in[24];
  const float* g2      = (const float*)d_in[25];
  const float* beta2   = (const float*)d_in[26];
  const float* g3      = (const float*)d_in[27];
  const float* beta3   = (const float*)d_in[28];

  // workspace carve (bytes)
  char* ws = (char*)d_ws;
  unsigned short* kT     = (unsigned short*)(ws);                // 67,108,864 B
  unsigned short* vT     = (unsigned short*)(ws + 67108864ull);  // 67,108,864 B
  float*          s_buf  = (float*)(ws + 134217728ull);          // 262,144 B
  float*          t_buf  = (float*)(ws + 134479872ull);          // 262,144 B
  float*          q_buf  = (float*)(ws + 134742016ull);          // 262,144 B
  float*          x_f32  = (float*)(ws + 135004160ull);          // 262,144 B
  unsigned short* x_bf16 = (unsigned short*)(ws + 135266304ull); // 131,072 B
  unsigned short* h1     = (unsigned short*)(ws + 135397376ull); // 2,097,152 B

  float* out_x    = (float*)d_out;             // (512,128)
  float* out_edge = out_x + 512 * 128;         // (512,512,128)

  k0_precompute<<<512, 128, 0, stream>>>(node, W_mem, Wq, bq, s_buf, t_buf, q_buf);

  ka_edge_fused<<<2048, 256, 103936, stream>>>(
      edge, W_mem /*We = first 128 rows*/, W_pe, Wk, Wv,
      b_mem, g_mem, be_mem, b_pe, g_pe, be_pe, g_ne, be_ne, bk, bv,
      s_buf, t_buf, kT, vT, out_edge);

  kb1_attention<<<512, 256, 0, stream>>>(q_buf, kT, vT, emask, Wo, bo, node,
                                         g2, beta2, x_f32, x_bf16);

  kb2a_ffn1<<<dim3(16, 4, 1), 256, 65536, stream>>>(x_bf16, W1, b1, h1);

  kb2b_ffn2<<<4, 256, 65536, stream>>>(h1, W2, b2, x_f32, g3, beta3, out_x);
}
